// GPT2LMHeadModel_50749333570205
// MI455X (gfx1250) — matrix operand order repaired and checked by reading, not compiled, not run on hardware
//
#include <hip/hip_runtime.h>
#include <stdint.h>

// ---------------- model constants (GPT-2 small, B=4, T=1024) ----------------
#define DMODEL 768
#define NH     12
#define HD64   64
#define NB     4
#define NT     1024
#define NL     12
#define NV     50257
#define MROWS  (NB*NT)          // 4096

typedef __attribute__((ext_vector_type(16))) __bf16 v16bf;
typedef __attribute__((ext_vector_type(8)))  float  v8f;

__device__ __forceinline__ float gelu_tanh(float x) {
  float x3 = x * x * x;
  return 0.5f * x * (1.0f + tanhf(0.7978845608028654f * (x + 0.044715f * x3)));
}

// ---------------- fp32 -> bf16 convert (no transpose; used for wte) ----------------
__global__ __launch_bounds__(256) void cvt_f32_bf16(const float* __restrict__ in,
                                                    __bf16* __restrict__ out,
                                                    long long n) {
  long long i = (long long)blockIdx.x * 256 + threadIdx.x;
  if (i < n) out[i] = (__bf16)in[i];
}

// ---------------- fp32 [L][K][N] -> bf16 [L][N][K] tiled transpose+convert ----------------
__global__ __launch_bounds__(256) void cvt_w_transpose(const float* __restrict__ in,
                                                       __bf16* __restrict__ out,
                                                       int K, int N) {
  __shared__ float tile[32][33];
  const int l  = blockIdx.z;
  const float* inl = in + (long long)l * K * N;
  __bf16*      outl = out + (long long)l * K * N;
  const int n0 = blockIdx.x * 32, k0 = blockIdx.y * 32;
  const int tx = threadIdx.x & 31, ty = threadIdx.x >> 5;   // 32 x 8
#pragma unroll
  for (int j = 0; j < 32; j += 8)
    tile[ty + j][tx] = inl[(long long)(k0 + ty + j) * N + (n0 + tx)];
  __syncthreads();
#pragma unroll
  for (int j = 0; j < 32; j += 8)
    outl[(long long)(n0 + ty + j) * K + (k0 + tx)] = (__bf16)tile[tx][ty + j];
}

// ---------------- V slice of qkv -> vT [z=b*H+h][64][T] (bf16, tiled) ----------------
__global__ __launch_bounds__(256) void transpose_v(const __bf16* __restrict__ qkv,
                                                   __bf16* __restrict__ vT) {
  __shared__ __bf16 tile[32][33];
  const int z = blockIdx.z;                    // 0..47
  const int b = z / NH, hh = z % NH;
  const int t0 = blockIdx.x * 32;              // along T
  const int n0 = blockIdx.y * 32;              // along 64
  const int tx = threadIdx.x & 31, ty = threadIdx.x >> 5;
  const __bf16* src = qkv + (long long)b * NT * 3 * DMODEL + 2 * DMODEL + hh * HD64;
#pragma unroll
  for (int j = 0; j < 32; j += 8)
    tile[ty + j][tx] = src[(long long)(t0 + ty + j) * 3 * DMODEL + (n0 + tx)];
  __syncthreads();
  __bf16* dst = vT + (long long)z * HD64 * NT;
#pragma unroll
  for (int j = 0; j < 32; j += 8)
    dst[(long long)(n0 + ty + j) * NT + (t0 + tx)] = tile[tx][ty + j];
}

// ---------------- token + position embedding ----------------
__global__ __launch_bounds__(256) void embed_kernel(const int* __restrict__ ids,
                                                    const float* __restrict__ wte,
                                                    const float* __restrict__ wpe,
                                                    float* __restrict__ x) {
  long long i = (long long)blockIdx.x * 256 + threadIdx.x;   // over B*T*D
  int d  = (int)(i % DMODEL);
  int bt = (int)(i / DMODEL);
  int t  = bt % NT;
  int id = ids[bt];
  x[i] = wte[(long long)id * DMODEL + d] + wpe[(long long)t * DMODEL + d];
}

// ---------------- LayerNorm (fp32 in) -> bf16 out, one block per row ----------------
__global__ __launch_bounds__(256) void layernorm_bf16(const float* __restrict__ x,
                                                      const float* __restrict__ g,
                                                      const float* __restrict__ b,
                                                      __bf16* __restrict__ out) {
  __shared__ float red[256];
  const int t = threadIdx.x;
  const long long row = blockIdx.x;
  const float* xr = x + row * DMODEL;
  float v0 = xr[t], v1 = xr[t + 256], v2 = xr[t + 512];
  red[t] = v0 + v1 + v2;
  __syncthreads();
  for (int off = 128; off > 0; off >>= 1) {
    if (t < off) red[t] += red[t + off];
    __syncthreads();
  }
  float mu = red[0] * (1.0f / DMODEL);
  __syncthreads();
  float d0 = v0 - mu, d1 = v1 - mu, d2 = v2 - mu;
  red[t] = d0 * d0 + d1 * d1 + d2 * d2;
  __syncthreads();
  for (int off = 128; off > 0; off >>= 1) {
    if (t < off) red[t] += red[t + off];
    __syncthreads();
  }
  float r = rsqrtf(red[0] * (1.0f / DMODEL) + 1e-5f);
  __bf16* orow = out + row * DMODEL;
  orow[t]       = (__bf16)(d0 * r * g[t]       + b[t]);
  orow[t + 256] = (__bf16)(d1 * r * g[t + 256] + b[t + 256]);
  orow[t + 512] = (__bf16)(d2 * r * g[t + 512] + b[t + 512]);
}

// ---------------- row softmax over 1024 cols (fp32 in, bf16 out) ----------------
__global__ __launch_bounds__(256) void softmax_rows(const float* __restrict__ a,
                                                    __bf16* __restrict__ o) {
  __shared__ float red[256];
  const int t = threadIdx.x;
  const long long row = blockIdx.x;
  const float* ar = a + row * NT;
  float v[4];
  float m = -3.0e38f;
#pragma unroll
  for (int k = 0; k < 4; ++k) { v[k] = ar[t + 256 * k]; m = fmaxf(m, v[k]); }
  red[t] = m;
  __syncthreads();
  for (int off = 128; off > 0; off >>= 1) {
    if (t < off) red[t] = fmaxf(red[t], red[t + off]);
    __syncthreads();
  }
  float mx = red[0];
  __syncthreads();
  float s = 0.f;
#pragma unroll
  for (int k = 0; k < 4; ++k) { v[k] = __expf(v[k] - mx); s += v[k]; }
  red[t] = s;
  __syncthreads();
  for (int off = 128; off > 0; off >>= 1) {
    if (t < off) red[t] += red[t + off];
    __syncthreads();
  }
  float inv = 1.0f / red[0];
  __bf16* orow = o + row * NT;
#pragma unroll
  for (int k = 0; k < 4; ++k) orow[t + 256 * k] = (__bf16)(v[k] * inv);
}

// ---------------- generic batched WMMA bf16 GEMM:  C = A @ B^T ----------------
// A: row-major [M][K] (lda), B: row-major [N][K] (ldb)  => C[m][n] = dot(A[m,:], B[n,:])
// Block tile 128x128, BK=64, double-buffered LDS filled with
// global_load_async_to_lds_b128 (ASYNCcnt), 8 waves of 4(M)x2(N), wave = 32x64.
// Batch offsets: off = (z/Hdiv)*s?o + (z%Hdiv)*s?i (element units).
union FragU { v16bf v; unsigned int u[8]; };

#define LDSTRIDE 72
#define TILEBYTES (128 * LDSTRIDE * 2)

template<bool BIAS, bool DOGELU, bool CAUSAL, bool RESID, bool OUTBF16>
__global__ __launch_bounds__(256)
void wmma_gemm(const __bf16* __restrict__ A, const __bf16* __restrict__ B,
               const float* __restrict__ bias, const float* __restrict__ Rsd,
               void* __restrict__ Cv, int M, int N, int K,
               int lda, int ldb, int ldc,
               long long sAo, long long sAi, long long sBo, long long sBi,
               long long sCo, long long sCi, int Hdiv, float alpha) {
  __shared__ __align__(16) __bf16 As[2][128 * LDSTRIDE];   // [m][k]
  __shared__ __align__(16) __bf16 Bs[2][128 * LDSTRIDE];   // [n][k]

  const int z = blockIdx.z;
  const long long offA = (long long)(z / Hdiv) * sAo + (long long)(z % Hdiv) * sAi;
  const long long offB = (long long)(z / Hdiv) * sBo + (long long)(z % Hdiv) * sBi;
  const long long offC = (long long)(z / Hdiv) * sCo + (long long)(z % Hdiv) * sCi;

  const int tid   = threadIdx.x;
  const int lane  = tid & 31;
  const int wid   = tid >> 5;
  const int waveM = wid & 3;    // 0..3  -> 32-row strip
  const int waveN = wid >> 2;   // 0..1  -> 64-col strip
  const int rowBase = blockIdx.y * 128;
  const int colBase = blockIdx.x * 128;

  const __bf16* Ab = A + offA;
  const __bf16* Bb = B + offB;

  const unsigned int asBase = (unsigned int)(uintptr_t)(&As[0][0]);
  const unsigned int bsBase = (unsigned int)(uintptr_t)(&Bs[0][0]);

  // per-thread tile-load geometry: 4 x b128 per matrix
  const int ldr = tid >> 3;             // 0..31, +32 per i
  const int ldk = (tid & 7) * 8;        // 0..56

  auto issue_tile = [&](int k0, int buf) {
#pragma unroll
    for (int i = 0; i < 4; ++i) {
      int r = ldr + 32 * i;
      const __bf16* g = Ab + (long long)(rowBase + r) * lda + (k0 + ldk);
      unsigned int laddr = asBase + (unsigned)buf * TILEBYTES + (unsigned)(r * LDSTRIDE + ldk) * 2;
      asm volatile("global_load_async_to_lds_b128 %0, %1, off"
                   :: "v"(laddr), "v"((unsigned long long)(uintptr_t)g) : "memory");
    }
#pragma unroll
    for (int i = 0; i < 4; ++i) {
      int n  = ldr + 32 * i;
      int gn = colBase + n;
      if (gn < N) {
        const __bf16* g = Bb + (long long)gn * ldb + (k0 + ldk);
        unsigned int laddr = bsBase + (unsigned)buf * TILEBYTES + (unsigned)(n * LDSTRIDE + ldk) * 2;
        asm volatile("global_load_async_to_lds_b128 %0, %1, off"
                     :: "v"(laddr), "v"((unsigned long long)(uintptr_t)g) : "memory");
      }
    }
  };

  v8f acc[2][4];
#pragma unroll
  for (int i = 0; i < 2; ++i)
#pragma unroll
    for (int j = 0; j < 4; ++j)
#pragma unroll
      for (int e = 0; e < 8; ++e) acc[i][j][e] = 0.0f;

  const int ktiles = K >> 6;
  issue_tile(0, 0);

  for (int kt = 0; kt < ktiles; ++kt) {
    const int cur = kt & 1;
    asm volatile("s_wait_asynccnt 0" ::: "memory");   // our tile writes to LDS done
    __syncthreads();                                  // everyone's writes visible; prev compute done
    if (kt + 1 < ktiles) issue_tile((kt + 1) << 6, cur ^ 1);

    const __bf16* Ac = &As[cur][0];
    const __bf16* Bc = &Bs[cur][0];
#pragma unroll
    for (int ks = 0; ks < 64; ks += 32) {
      FragU fa[2], fb[4];
      const int mrow  = lane & 15;
      const int khalf = (lane >> 4) * 8;    // A: lanes16-31 hold K+8 group
      const int kh2   = (lane >> 4) * 16;   // unused; B also reads with khalf, same k order as A
#pragma unroll
      for (int i = 0; i < 2; ++i) {
        int r = waveM * 32 + i * 16 + mrow;
#pragma unroll
        for (int p = 0; p < 8; ++p) {
          int kk = ks + ((p & 4) ? 16 : 0) + khalf + (p & 3) * 2;
          fa[i].u[p] = *(const unsigned int*)(&Ac[r * LDSTRIDE + kk]);
        }
      }
#pragma unroll
      for (int j = 0; j < 4; ++j) {
        int n = waveN * 64 + j * 16 + (lane & 15);
#pragma unroll
        for (int p = 0; p < 8; ++p) {
          int kk = ks + ((p & 4) ? 16 : 0) + khalf + (p & 3) * 2;
          fb[j].u[p] = *(const unsigned int*)(&Bc[n * LDSTRIDE + kk]);
        }
      }
#pragma unroll
      for (int i = 0; i < 2; ++i)
#pragma unroll
        for (int j = 0; j < 4; ++j)
          acc[i][j] = __builtin_amdgcn_wmma_f32_16x16x32_bf16(
              false, fa[i].v, false, fb[j].v, (short)0, acc[i][j], false, false);
    }
  }

  // ---- epilogue: C layout: VGPR e -> M = e (lanes 0-15) / e+8 (lanes 16-31), N = lane&15 ----
  const int nloc = lane & 15;
  const int mhi  = (lane >> 4) * 8;
  float*  Cf = (float*)Cv;
  __bf16* Cb = (__bf16*)Cv;
#pragma unroll
  for (int i = 0; i < 2; ++i) {
#pragma unroll
    for (int j = 0; j < 4; ++j) {
      int col = colBase + waveN * 64 + j * 16 + nloc;
      if (col < N) {
#pragma unroll
        for (int e = 0; e < 8; ++e) {
          int row = rowBase + waveM * 32 + i * 16 + e + mhi;
          float v = acc[i][j][e] * alpha;
          if (CAUSAL) { if (col > row) v = -1.0e30f; }
          if (BIAS)   v += bias[col];
          if (DOGELU) v = gelu_tanh(v);
          long long idx = offC + (long long)row * ldc + col;
          if (RESID)  v += Rsd[idx];
          if (OUTBF16) Cb[idx] = (__bf16)v; else Cf[idx] = v;
        }
      }
    }
  }
}

// ---------------- host side ----------------
static inline size_t alignup256(size_t v) { return (v + 255) & ~(size_t)255; }

extern "C" void kernel_launch(void* const* d_in, const int* in_sizes, int n_in,
                              void* d_out, int out_size, void* d_ws, size_t ws_size,
                              hipStream_t stream) {
  (void)in_sizes; (void)n_in; (void)out_size; (void)ws_size;
  const int*   ids     = (const int*)  d_in[0];
  const float* wte     = (const float*)d_in[1];
  const float* wpe     = (const float*)d_in[2];
  const float* ln1_g   = (const float*)d_in[3];
  const float* ln1_b   = (const float*)d_in[4];
  const float* w_attn  = (const float*)d_in[5];
  const float* b_attn  = (const float*)d_in[6];
  const float* w_proj  = (const float*)d_in[7];
  const float* b_proj  = (const float*)d_in[8];
  const float* ln2_g   = (const float*)d_in[9];
  const float* ln2_b   = (const float*)d_in[10];
  const float* w_fc    = (const float*)d_in[11];
  const float* b_fc    = (const float*)d_in[12];
  const float* w_mproj = (const float*)d_in[13];
  const float* b_mproj = (const float*)d_in[14];
  const float* lnf_g   = (const float*)d_in[15];
  const float* lnf_b   = (const float*)d_in[16];

  // ---- workspace carve (~640 MB) ----
  char* p = (char*)d_ws;
  auto take = [&](size_t bytes) { void* r = (void*)p; p += alignup256(bytes); return r; };
  __bf16* wteB = (__bf16*)take((size_t)NV * DMODEL * 2);                       // [V][768]
  __bf16* waT  = (__bf16*)take((size_t)NL * DMODEL * 3 * DMODEL * 2);          // [l][2304][768]
  __bf16* wpT  = (__bf16*)take((size_t)NL * DMODEL * DMODEL * 2);              // [l][768][768]
  __bf16* wfT  = (__bf16*)take((size_t)NL * DMODEL * 4 * DMODEL * 2);          // [l][3072][768]
  __bf16* wmT  = (__bf16*)take((size_t)NL * 4 * DMODEL * DMODEL * 2);          // [l][768][3072]
  float*  x    = (float*) take((size_t)MROWS * DMODEL * 4);
  __bf16* h    = (__bf16*)take((size_t)MROWS * DMODEL * 2);
  __bf16* qkvB = (__bf16*)take((size_t)MROWS * 3 * DMODEL * 2);
  __bf16* vT   = (__bf16*)take((size_t)NB * NH * HD64 * NT * 2);               // [z][64][1024]
  float*  attF = (float*) take((size_t)NB * NH * NT * NT * 4);
  __bf16* attB = (__bf16*)take((size_t)NB * NH * NT * NT * 2);
  __bf16* oB   = (__bf16*)take((size_t)MROWS * DMODEL * 2);
  __bf16* m1   = (__bf16*)take((size_t)MROWS * 4 * DMODEL * 2);

  dim3 blk(256);
  // wte -> bf16 (row-major already matches B^T layout for lm_head)
  {
    long long n = (long long)NV * DMODEL;
    cvt_f32_bf16<<<dim3((unsigned)((n + 255) / 256)), blk, 0, stream>>>(wte, wteB, n);
  }
  // weights -> bf16 transposed [N][K]
  cvt_w_transpose<<<dim3(3 * DMODEL / 32, DMODEL / 32, NL), blk, 0, stream>>>(w_attn,  waT, DMODEL, 3 * DMODEL);
  cvt_w_transpose<<<dim3(DMODEL / 32,     DMODEL / 32, NL), blk, 0, stream>>>(w_proj,  wpT, DMODEL, DMODEL);
  cvt_w_transpose<<<dim3(4 * DMODEL / 32, DMODEL / 32, NL), blk, 0, stream>>>(w_fc,    wfT, DMODEL, 4 * DMODEL);
  cvt_w_transpose<<<dim3(DMODEL / 32, 4 * DMODEL / 32, NL), blk, 0, stream>>>(w_mproj, wmT, 4 * DMODEL, DMODEL);

  // x = wte[ids] + wpe
  embed_kernel<<<dim3(MROWS * DMODEL / 256), blk, 0, stream>>>(ids, wte, wpe, x);

  const long long sQKo = (long long)NT * 3 * DMODEL;   // per-batch stride in qkv
  const long long sATo = (long long)NH * NT * NT;      // per-batch stride in att
  const long long sATi = (long long)NT * NT;           // per-head stride in att
  const long long sOo  = (long long)NT * DMODEL;       // per-batch stride in o/x
  const long long sVTo = (long long)NH * HD64 * NT;    // per-batch stride in vT
  const long long sVTi = (long long)HD64 * NT;         // per-head stride in vT

  for (int l = 0; l < NL; ++l) {
    // h = LN1(x)
    layernorm_bf16<<<dim3(MROWS), blk, 0, stream>>>(x, ln1_g + l * DMODEL, ln1_b + l * DMODEL, h);

    // qkv = h @ w_attn + b_attn     [4096 x 2304], B^T = waT rows
    wmma_gemm<true, false, false, false, true>
        <<<dim3(3 * DMODEL / 128, MROWS / 128, 1), blk, 0, stream>>>(
        h, waT + (size_t)l * DMODEL * 3 * DMODEL, b_attn + (size_t)l * 3 * DMODEL, nullptr,
        qkvB, MROWS, 3 * DMODEL, DMODEL, DMODEL, DMODEL, 3 * DMODEL,
        0, 0, 0, 0, 0, 0, 1, 1.0f);

    // vT[z] = V^T
    transpose_v<<<dim3(NT / 32, HD64 / 32, NB * NH), blk, 0, stream>>>(qkvB, vT);

    // scores = scale * Q @ K^T, causal  [48 x 1024 x 1024], K=64
    wmma_gemm<false, false, true, false, false>
        <<<dim3(NT / 128, NT / 128, NB * NH), blk, 0, stream>>>(
        qkvB, qkvB + DMODEL, nullptr, nullptr, attF,
        NT, NT, HD64, 3 * DMODEL, 3 * DMODEL, NT,
        sQKo, HD64, sQKo, HD64, sATo, sATi, NH, 0.125f);

    // softmax rows -> bf16
    softmax_rows<<<dim3(NB * NH * NT), blk, 0, stream>>>(attF, attB);

    // o = P @ V = P @ (vT)^T   [48 batches: 1024 x 64, K=1024] -> packed into [B,T,D]
    wmma_gemm<false, false, false, false, true>
        <<<dim3(1, NT / 128, NB * NH), blk, 0, stream>>>(
        attB, vT, nullptr, nullptr, oB,
        NT, HD64, NT, NT, NT, DMODEL,
        sATo, sATi, sVTo, sVTi, sOo, HD64, NH, 1.0f);

    // x = x + o @ w_proj + b_proj
    wmma_gemm<true, false, false, true, false>
        <<<dim3(DMODEL / 128, MROWS / 128, 1), blk, 0, stream>>>(
        oB, wpT + (size_t)l * DMODEL * DMODEL, b_proj + (size_t)l * DMODEL, x,
        x, MROWS, DMODEL, DMODEL, DMODEL, DMODEL, DMODEL,
        0, 0, 0, 0, 0, 0, 1, 1.0f);

    // h = LN2(x)
    layernorm_bf16<<<dim3(MROWS), blk, 0, stream>>>(x, ln2_g + l * DMODEL, ln2_b + l * DMODEL, h);

    // m1 = gelu(h @ w_fc + b_fc)    [4096 x 3072]
    wmma_gemm<true, true, false, false, true>
        <<<dim3(4 * DMODEL / 128, MROWS / 128, 1), blk, 0, stream>>>(
        h, wfT + (size_t)l * DMODEL * 4 * DMODEL, b_fc + (size_t)l * 4 * DMODEL, nullptr,
        m1, MROWS, 4 * DMODEL, DMODEL, DMODEL, DMODEL, 4 * DMODEL,
        0, 0, 0, 0, 0, 0, 1, 1.0f);

    // x = x + m1 @ w_mproj + b_mproj   [K=3072]
    wmma_gemm<true, false, false, true, false>
        <<<dim3(DMODEL / 128, MROWS / 128, 1), blk, 0, stream>>>(
        m1, wmT + (size_t)l * 4 * DMODEL * DMODEL, b_mproj + (size_t)l * DMODEL, x,
        x, MROWS, DMODEL, 4 * DMODEL, 4 * DMODEL, 4 * DMODEL, DMODEL,
        0, 0, 0, 0, 0, 0, 1, 1.0f);
  }

  // h = LN_f(x)
  layernorm_bf16<<<dim3(MROWS), blk, 0, stream>>>(x, lnf_g, lnf_b, h);

  // logits = h @ wte^T   [4096 x 50257], fp32 out
  wmma_gemm<false, false, false, false, false>
      <<<dim3((NV + 127) / 128, MROWS / 128, 1), blk, 0, stream>>>(
      h, wteB, nullptr, nullptr, d_out,
      MROWS, NV, DMODEL, DMODEL, DMODEL, NV,
      0, 0, 0, 0, 0, 0, 1, 1.0f);
}
